// CIN_42666205119104
// MI455X (gfx1250) — matrix operand order repaired and checked by reading, not compiled, not run on hardware
//
#include <hip/hip_runtime.h>

// ---------------------------------------------------------------------------
// CIN (xDeepFM) stack on gfx1250. Per layer:
//   C[256 x (B*D)] = W[256 x cin] @ X[cin x (B*D)],
//   X[(h*F+f),(b,d)] = hidden[b,h,d] * x0[b,f,d]   (built on the fly in LDS)
// v_wmma_f32_16x16x32_bf16, 2 batch elements per workgroup, 128-K chunks.
// ---------------------------------------------------------------------------

typedef __attribute__((ext_vector_type(16))) __bf16        v16bf;
typedef __attribute__((ext_vector_type(8)))  float         v8f;
typedef __attribute__((ext_vector_type(4)))  unsigned int  u32x4;

// ----------------------- conversion kernels --------------------------------

__global__ __launch_bounds__(256) void cvt_f32_bf16(const float* __restrict__ src,
                                                    __bf16* __restrict__ dst, int n) {
  int i = blockIdx.x * 256 + threadIdx.x;
  if (i < n) dst[i] = (__bf16)src[i];
}

// w0 is [256][1521]; pad K to 1536 with zeros so the WMMA K-loop is uniform.
__global__ __launch_bounds__(256) void cvt_w0_pad(const float* __restrict__ src,
                                                  __bf16* __restrict__ dst) {
  int i = blockIdx.x * 256 + threadIdx.x;
  if (i < 256 * 1536) {
    int o = i / 1536;
    int c = i - o * 1536;
    dst[i] = (c < 1521) ? (__bf16)src[o * 1521 + c] : (__bf16)0.0f;
  }
}

// ----------------------- fused CIN layer kernel ----------------------------
// One workgroup per 2 batch elements. 256 threads = 8 waves.
// Wave w computes output rows [32w, 32w+32) x 64 columns (2 batches x 32 d).
template <int H, int CIN, int CINP, int DIRECT, bool WRITE_HIDDEN>
__global__ __launch_bounds__(256) void cin_layer(
    const __bf16* __restrict__ x0b,     // [B][39][32] bf16
    const __bf16* __restrict__ hid_in,  // [B][H][32]  bf16
    const __bf16* __restrict__ wp,      // [256][CINP] bf16 (zero-padded)
    const float*  __restrict__ bias,    // [256] f32
    float*        __restrict__ out,     // [B][512] f32
    int out_off,
    __bf16*       __restrict__ hid_out) // [B][128][32] bf16 (or null)
{
  constexpr int F  = 39, D = 32;
  constexpr int KC = 128;               // K per LDS chunk (4 WMMA K-steps)
  constexpr int KS = 136;               // padded LDS row stride (272 B, b64-ok)

  __shared__ __bf16 xs[2][F * D];
  __shared__ __bf16 hs[2][128 * D];
  __shared__ __align__(16) __bf16 Xs[64 * KS];   // [col][k], col = bsel*32 + d
  __shared__ float bias_s[256];

  const int tid = threadIdx.x;
  const int b0  = 2 * blockIdx.x;
  const int b1  = b0 + 1;

  // Cooperative dword loads of the per-batch tiles.
  #pragma unroll
  for (int bb = 0; bb < 2; ++bb) {
    const int b = b0 + bb;
    {
      const unsigned int* src = (const unsigned int*)(x0b + (size_t)b * (F * D));
      unsigned int*       dst = (unsigned int*)&xs[bb][0];
      for (int i = tid; i < F * D / 2; i += 256) dst[i] = src[i];
    }
    {
      const unsigned int* src = (const unsigned int*)(hid_in + (size_t)b * (H * D));
      unsigned int*       dst = (unsigned int*)&hs[bb][0];
      for (int i = tid; i < H * D / 2; i += 256) dst[i] = src[i];
    }
  }
  bias_s[tid] = bias[tid];

  const int lane  = tid & 31;
  const int wave  = tid >> 5;
  const int n     = lane & 15;   // tile column / A-row index
  const int half  = lane >> 4;   // K-half for A/B fragments, M-half for C
  const int mbase = wave * 32;

  v8f acc[2][4] = {};

  // X-chunk build assignment: thread covers kk_base+32i (i<4), col groups
  // (tid&7)*2+j (j<2), each group = 4 consecutive columns.
  const int kk_base = tid >> 3;        // 0..31
  const int gbase   = (tid & 7) * 2;   // 0,2,..,14

  // Incremental h = c/39, f = c%39 tracking (c advances by KC=128 per chunk;
  // 128 = 3*39 + 11).
  int hh[4], ff[4];
  #pragma unroll
  for (int i = 0; i < 4; ++i) {
    int c = kk_base + 32 * i;
    hh[i] = c / F;
    ff[i] = c - F * hh[i];
  }

  for (int kb = 0; kb < CINP; kb += KC) {
    __syncthreads();  // previous chunk fully consumed (also covers tile loads)

    // Prefetch next weight block while we build the X chunk.
    if (kb + KC < CINP) {
      __builtin_prefetch(wp + (size_t)(mbase + n) * CINP + kb + KC, 0, 1);
      __builtin_prefetch(wp + (size_t)(mbase + 16 + n) * CINP + kb + KC, 0, 1);
    }

    #pragma unroll
    for (int i = 0; i < 4; ++i) {
      const int kk = kk_base + 32 * i;
      const int c  = kb + kk;
      const int h  = hh[i];
      const int f  = ff[i];
      #pragma unroll
      for (int j = 0; j < 2; ++j) {
        const int grp  = gbase + j;        // 0..15
        const int bsel = grp >> 3;
        const int dd   = (grp & 7) * 4;    // 0,4,...,28
        if (c < CIN) {
          const __bf16* hr = &hs[bsel][h * D + dd];
          const __bf16* xr = &xs[bsel][f * D + dd];
          #pragma unroll
          for (int jj = 0; jj < 4; ++jj) {
            float v = (float)hr[jj] * (float)xr[jj];
            Xs[(grp * 4 + jj) * KS + kk] = (__bf16)v;
          }
        } else {
          #pragma unroll
          for (int jj = 0; jj < 4; ++jj)
            Xs[(grp * 4 + jj) * KS + kk] = (__bf16)0.0f;
        }
      }
      // advance (h,f) for c += 128
      int fn = ff[i] + 11;
      int hn = hh[i] + 3;
      if (fn >= F) { fn -= F; hn += 1; }
      ff[i] = fn;
      hh[i] = hn;
    }
    __syncthreads();  // chunk visible to all waves

    #pragma unroll
    for (int ks = 0; ks < 4; ++ks) {
      const int k0 = kb + ks * 32;

      // A fragments (weights, bf16 16x32 layout): two contiguous b128 per
      // lane. Lanes 0-15: K k0+0..15 ; lanes 16-31: K k0+16..31.
      union AF { u32x4 u[2]; v16bf v; };
      AF a[2];
      #pragma unroll
      for (int mi = 0; mi < 2; ++mi) {
        const u32x4* pa =
            (const u32x4*)(wp + (size_t)(mbase + mi * 16 + n) * CINP + k0 + 16 * half);
        a[mi].u[0] = pa[0];
        a[mi].u[1] = pa[1];
      }

      // B fragments from LDS: lane holds column ni*16+n, K = 16*half + 0..15.
      union BF { unsigned long long q[4]; v16bf v; };
      BF bf[4];
      #pragma unroll
      for (int ni = 0; ni < 4; ++ni) {
        const unsigned long long* q =
            (const unsigned long long*)(Xs + (ni * 16 + n) * KS + ks * 32 + half * 16);
        bf[ni].q[0] = q[0]; bf[ni].q[1] = q[1];
        bf[ni].q[2] = q[2]; bf[ni].q[3] = q[3];
      }

      #pragma unroll
      for (int mi = 0; mi < 2; ++mi)
        #pragma unroll
        for (int ni = 0; ni < 4; ++ni)
          acc[mi][ni] = __builtin_amdgcn_wmma_f32_16x16x32_bf16(
              false, a[mi].v, false, bf[ni].v, (short)0, acc[mi][ni], false, false);
    }
  }

  // Epilogue: bias + ReLU, bf16 hidden store, and sum over D via shfl_xor
  // butterfly across the 16 lanes of each half (wave32-safe).
  #pragma unroll
  for (int mi = 0; mi < 2; ++mi) {
    #pragma unroll
    for (int r = 0; r < 8; ++r) {
      const int   o  = mbase + mi * 16 + r + 8 * half;  // C layout: M = r + 8*half
      const float bb = bias_s[o];
      float v[4];
      #pragma unroll
      for (int ni = 0; ni < 4; ++ni) v[ni] = fmaxf(acc[mi][ni][r] + bb, 0.0f);

      if (WRITE_HIDDEN && o >= DIRECT) {
        __bf16* h0p = hid_out + ((size_t)b0 * 128 + (o - DIRECT)) * D;
        h0p[n]      = (__bf16)v[0];
        h0p[16 + n] = (__bf16)v[1];
        __bf16* h1p = hid_out + ((size_t)b1 * 128 + (o - DIRECT)) * D;
        h1p[n]      = (__bf16)v[2];
        h1p[16 + n] = (__bf16)v[3];
      }

      float s0 = v[0] + v[1];
      float s1 = v[2] + v[3];
      s0 += __shfl_xor(s0, 1); s1 += __shfl_xor(s1, 1);
      s0 += __shfl_xor(s0, 2); s1 += __shfl_xor(s1, 2);
      s0 += __shfl_xor(s0, 4); s1 += __shfl_xor(s1, 4);
      s0 += __shfl_xor(s0, 8); s1 += __shfl_xor(s1, 8);
      if (n == 0 && o < DIRECT) {
        out[(size_t)b0 * 512 + out_off + o] = s0;
        out[(size_t)b1 * 512 + out_off + o] = s1;
      }
    }
  }
}

// ----------------------------- launch --------------------------------------

extern "C" void kernel_launch(void* const* d_in, const int* in_sizes, int n_in,
                              void* d_out, int out_size, void* d_ws, size_t ws_size,
                              hipStream_t stream) {
  (void)in_sizes; (void)n_in; (void)out_size; (void)ws_size;

  const float* x  = (const float*)d_in[0];
  const float* w0 = (const float*)d_in[1];
  const float* b0 = (const float*)d_in[2];
  const float* w1 = (const float*)d_in[3];
  const float* b1 = (const float*)d_in[4];
  const float* w2 = (const float*)d_in[5];
  const float* b2 = (const float*)d_in[6];
  float* out = (float*)d_out;

  constexpr int BATCH = 2048, F = 39, D = 32;
  constexpr int XN  = BATCH * F * D;     // 2,555,904
  constexpr int W0N = 256 * 1536;        // padded w0
  constexpr int W1N = 256 * 4992;
  constexpr size_t HBYTES = (size_t)BATCH * 128 * D * 2;

  char* p = (char*)d_ws;
  __bf16* x0b = (__bf16*)p;  p += (size_t)XN  * 2;   // 5,111,808 B (16B aligned)
  __bf16* w0b = (__bf16*)p;  p += (size_t)W0N * 2;   //   786,432 B
  __bf16* w1b = (__bf16*)p;  p += (size_t)W1N * 2;   // 2,555,904 B
  __bf16* w2b = (__bf16*)p;  p += (size_t)W1N * 2;
  __bf16* h1  = (__bf16*)p;  p += HBYTES;            // 16 MiB
  __bf16* h2  = (__bf16*)p;                          // 16 MiB

  cvt_f32_bf16<<<(XN  + 255) / 256, 256, 0, stream>>>(x,  x0b, XN);
  cvt_w0_pad  <<<(W0N + 255) / 256, 256, 0, stream>>>(w0, w0b);
  cvt_f32_bf16<<<(W1N + 255) / 256, 256, 0, stream>>>(w1, w1b, W1N);
  cvt_f32_bf16<<<(W1N + 255) / 256, 256, 0, stream>>>(w2, w2b, W1N);

  // Layer 0: hidden = x0 (H=39), cin = 39*39 = 1521 -> padded 1536 = 12*128.
  cin_layer<39, 1521, 1536, 128, true>
      <<<BATCH / 2, 256, 0, stream>>>(x0b, x0b, w0b, b0, out, 0, h1);
  // Layer 1: H=128, cin = 128*39 = 4992 = 39*128.
  cin_layer<128, 4992, 4992, 128, true>
      <<<BATCH / 2, 256, 0, stream>>>(x0b, h1, w1b, b1, out, 128, h2);
  // Layer 2: H=128, cin = 4992, all 256 rows are "direct", no hidden out.
  cin_layer<128, 4992, 4992, 256, false>
      <<<BATCH / 2, 256, 0, stream>>>(x0b, h2, w2b, b2, out, 256, (__bf16*)nullptr);
}
